// Conv2DLinalgRMSNorm_36318243455328
// MI455X (gfx1250) — matrix operand order repaired and checked by reading, not compiled, not run on hardware
//
#include <hip/hip_runtime.h>
#include <hip/hip_bf16.h>

#define HIDDEN   1024
#define NLAYERS  20
#define MTOTAL   8192                 // 2 * 4096 rows
#define EPSH     1.024e-3f            // EPS * HIDDEN = 1e-6 * 1024
#define SQRTH    32.0f                // sqrt(HIDDEN)
#define KSTEP    32

typedef __attribute__((ext_vector_type(16))) __bf16       v16bf;
typedef __attribute__((ext_vector_type(8)))  float        v8f;
typedef __attribute__((ext_vector_type(4)))  unsigned int u32x4;
typedef __attribute__((ext_vector_type(8)))  int          i32x8;
typedef __attribute__((ext_vector_type(4)))  int          i32x4;

// LDS tile sizes (bytes)
#define BTILE_BYTES (HIDDEN * KSTEP * 2)   // 64 KB : B tile [1024 rows][32 k]
#define ATILE_BYTES (32 * KSTEP * 2)       //  2 KB : A tile [32 rows][32 k]
#define BUF_BYTES   (BTILE_BYTES + ATILE_BYTES)
#define NBUF        2

#if defined(__has_builtin)
#if __has_builtin(__builtin_amdgcn_tensor_load_to_lds) && \
    __has_builtin(__builtin_amdgcn_s_wait_tensorcnt)
#define HAVE_TDM 1
#endif
#endif
#ifndef HAVE_TDM
#define HAVE_TDM 0
#endif

// ---------------------------------------------------------------------------
// Kernel 1: w_bf16[n][k] = (sum_l conv_w[l][n][k]) / 20, stored bf16 row-major.
// ---------------------------------------------------------------------------
__global__ void wsum_to_bf16(const float* __restrict__ w, __bf16* __restrict__ wb) {
    int idx = blockIdx.x * blockDim.x + threadIdx.x;          // float4 index
    const float4* p = (const float4*)w + idx;
    float4 s = p[0];
#pragma unroll
    for (int l = 1; l < NLAYERS; ++l) {
        float4 v = p[(size_t)l * (HIDDEN * HIDDEN / 4)];
        s.x += v.x; s.y += v.y; s.z += v.z; s.w += v.w;
    }
    const float inv = 1.0f / (float)NLAYERS;                  // fold the /20 here
    union { __bf16 h[4]; uint2 u; } o;
    o.h[0] = (__bf16)(s.x * inv);
    o.h[1] = (__bf16)(s.y * inv);
    o.h[2] = (__bf16)(s.z * inv);
    o.h[3] = (__bf16)(s.w * inv);
    ((uint2*)wb)[idx] = o.u;
}

// ---------------------------------------------------------------------------
// Kernel 2: x_bf16 = bf16(x)
// ---------------------------------------------------------------------------
__global__ void x_to_bf16(const float* __restrict__ x, __bf16* __restrict__ xb) {
    size_t idx = (size_t)blockIdx.x * blockDim.x + threadIdx.x;
    float4 v = ((const float4*)x)[idx];
    union { __bf16 h[4]; uint2 u; } o;
    o.h[0] = (__bf16)v.x; o.h[1] = (__bf16)v.y;
    o.h[2] = (__bf16)v.z; o.h[3] = (__bf16)v.w;
    ((uint2*)xb)[idx] = o.u;
}

// ---------------------------------------------------------------------------
// TDM 2D tile load: build D# (groups 0/1 per CDNA5 ISA)
// and issue TENSOR_LOAD_TO_LDS. data_size = 2 bytes (bf16).
// Toolchain here exposes the 6-arg builtin:
//   (u32x4 g0, i32x8 g1, i32x4 g2, i32x4 g3, i32x8 extra, i32 cpol)
// ---------------------------------------------------------------------------
#if HAVE_TDM
__device__ __forceinline__ void tdm_load_2d(
    unsigned long long gaddr,      // global byte address of tile start
    unsigned int lds_addr,         // LDS byte address of tile dest
    unsigned int tensor_d0,        // tensor dim0 length (elements)
    unsigned int tensor_d1,        // tensor dim1 length (rows)
    unsigned long long d0_stride,  // stride between rows (elements)
    unsigned int tile_d0,          // tile dim0 (elements, contiguous)
    unsigned int tile_d1)          // tile dim1 (rows)
{
    u32x4 g0;
    g0[0] = 1u;                                            // count=1, user mode
    g0[1] = lds_addr;                                      // lds_addr [63:32]
    g0[2] = (unsigned int)gaddr;                           // global_addr lo
    g0[3] = ((unsigned int)(gaddr >> 32) & 0x01FFFFFFu)    // global_addr [56:32]
            | (2u << 30);                                  // type=2 (image)
    i32x8 g1;
    g1[0] = (int)(1u << 16);                               // data_size=1 (2B)
    g1[1] = (int)((tensor_d0 & 0xFFFFu) << 16);            // tensor_dim0 lo16
    g1[2] = (int)((tensor_d0 >> 16) | ((tensor_d1 & 0xFFFFu) << 16));
    g1[3] = (int)((tensor_d1 >> 16) | ((tile_d0 & 0xFFFFu) << 16));
    g1[4] = (int)(tile_d1 & 0xFFFFu);                      // tile_dim2 = 0
    g1[5] = (int)(unsigned int)d0_stride;                  // dim0_stride lo32
    g1[6] = (int)((unsigned int)(d0_stride >> 32) & 0xFFFFu); // hi16; dim1_stride=0
    g1[7] = 0;
    i32x4 z4 = {0, 0, 0, 0};
    i32x8 z8 = {0, 0, 0, 0, 0, 0, 0, 0};
    __builtin_amdgcn_tensor_load_to_lds(g0, g1, z4, z4, z8, 0);
}
#endif

// ---------------------------------------------------------------------------
// Kernel 3: fused GEMM (bf16 WMMA, f32 acc) + RMSNorm epilogue.
// Block = 256 threads = 8 waves; wave w owns N columns [w*128, w*128+128).
// Block tile = 32 rows x full N=1024. TDM double-buffers A/B tiles in LDS.
// ---------------------------------------------------------------------------
__global__ __launch_bounds__(256) void gemm_rms(
    const __bf16* __restrict__ xb, const __bf16* __restrict__ wb,
    const float* __restrict__ norm_w, float* __restrict__ out)
{
    extern __shared__ char smem[];     // NBUF * (B tile + A tile)
    __shared__ float red[32];

    const int wave = threadIdx.x >> 5;
    const int lane = threadIdx.x & 31;
    const int half = lane >> 4;        // which 16-lane half
    const int r    = lane & 15;
    const int wn   = wave;             // 128-wide n column (0..7)
    const int mblk = blockIdx.x * 32;
    const int nsteps = HIDDEN / KSTEP; // 32

#if HAVE_TDM
    // issue one k-step's DMA pair (B tile + A tile) into buffer kb%NBUF
    auto stage = [&](int kb) {
        const unsigned int lbase =
            (unsigned int)(size_t)(smem + (kb % NBUF) * BUF_BYTES);
        tdm_load_2d((unsigned long long)(size_t)wb + (size_t)kb * KSTEP * 2,
                    lbase, HIDDEN, HIDDEN, HIDDEN, KSTEP, HIDDEN);       // B
        tdm_load_2d((unsigned long long)(size_t)xb +
                        ((size_t)mblk * HIDDEN + (size_t)kb * KSTEP) * 2,
                    lbase + BTILE_BYTES, HIDDEN, MTOTAL, HIDDEN, KSTEP, 32); // A
    };
    if (wave == 0) stage(0);
#else
    // synchronous cooperative staging fallback (single buffer)
    auto stage_sync = [&](int kb) {
        char* lbase = smem;
        const char* gB = (const char*)wb + (size_t)kb * KSTEP * 2;
#pragma unroll
        for (int i = 0; i < 16; ++i) {                 // 4096 16B chunks
            int c = threadIdx.x + i * 256;
            int row = c >> 2, sub = c & 3;
            *(uint4*)(lbase + row * 64 + sub * 16) =
                *(const uint4*)(gB + (size_t)row * HIDDEN * 2 + sub * 16);
        }
        const char* gA = (const char*)xb +
                         ((size_t)mblk * HIDDEN + (size_t)kb * KSTEP) * 2;
        if (threadIdx.x < 128) {                       // 128 16B chunks
            int c = threadIdx.x;
            int row = c >> 2, sub = c & 3;
            *(uint4*)(lbase + BTILE_BYTES + row * 64 + sub * 16) =
                *(const uint4*)(gA + (size_t)row * HIDDEN * 2 + sub * 16);
        }
    };
#endif

    v8f acc[8][2];
    const v8f vzero = {0.f, 0.f, 0.f, 0.f, 0.f, 0.f, 0.f, 0.f};
#pragma unroll
    for (int t = 0; t < 8; ++t) { acc[t][0] = vzero; acc[t][1] = vzero; }

    for (int kb = 0; kb < nsteps; ++kb) {
#if HAVE_TDM
        if (wave == 0) __builtin_amdgcn_s_wait_tensorcnt((short)0);
        __syncthreads();                       // buffer kb ready for everyone
        if (wave == 0 && kb + 1 < nsteps) stage(kb + 1);
        const char* lb = smem + (kb % NBUF) * BUF_BYTES;
#else
        if (kb) __syncthreads();               // everyone done with previous
        stage_sync(kb);
        __syncthreads();
        const char* lb = smem;
#endif
        const char* la = lb + BTILE_BYTES;

        // ---- A fragments from LDS (16x32 bf16, ISA layout):
        // lanes 0-15: local K {0..7, 16..23}; lanes 16-31: {8..15, 24..31}
        union { v16bf v; uint4 q[2]; } a0, a1;
        const char* a0p = la + (0  + r) * 64 + half * 16;
        const char* a1p = la + (16 + r) * 64 + half * 16;
        a0.q[0] = *(const uint4*)(a0p);
        a0.q[1] = *(const uint4*)(a0p + 32);
        a1.q[0] = *(const uint4*)(a1p);
        a1.q[1] = *(const uint4*)(a1p + 32);

#pragma unroll
        for (int t = 0; t < 8; ++t) {
            // ---- B fragment (32x16): lane = column n; halves take local
            // K {0..7, 16..23} / {8..15, 24..31} (same order as A).
            const char* bp = lb + (wn * 128 + t * 16 + r) * 64 + half * 16;
            union { v16bf v; uint4 q[2]; } b;
            b.q[0] = *(const uint4*)(bp);
            b.q[1] = *(const uint4*)(bp + 32);
            acc[t][0] = __builtin_amdgcn_wmma_f32_16x16x32_bf16(
                false, a0.v, false, b.v, (short)0, acc[t][0], false, false);
            acc[t][1] = __builtin_amdgcn_wmma_f32_16x16x32_bf16(
                false, a1.v, false, b.v, (short)0, acc[t][1], false, false);
        }
    }

    // ---------------- RMSNorm epilogue ----------------
    __syncthreads();
    if (threadIdx.x < 32) red[threadIdx.x] = 0.0f;
    __syncthreads();

    // C layout: VGPR v -> row v (lanes 0-15), row v+8 (lanes 16-31)
#pragma unroll
    for (int mt = 0; mt < 2; ++mt) {
#pragma unroll
        for (int v = 0; v < 8; ++v) {
            float p = 0.f;
#pragma unroll
            for (int t = 0; t < 8; ++t) p += acc[t][mt][v] * acc[t][mt][v];
            p += __shfl_xor(p, 8, 16);
            p += __shfl_xor(p, 4, 16);
            p += __shfl_xor(p, 2, 16);
            p += __shfl_xor(p, 1, 16);
            if (r == 0) atomicAdd(&red[mt * 16 + v + half * 8], p);
        }
    }
    __syncthreads();

    if (threadIdx.x < 32)
        red[threadIdx.x] = SQRTH * rsqrtf(red[threadIdx.x] + EPSH);
    __syncthreads();

#pragma unroll
    for (int mt = 0; mt < 2; ++mt) {
#pragma unroll
        for (int v = 0; v < 8; ++v) {
            const int   mloc = mt * 16 + v + half * 8;
            const float s    = red[mloc];
            const size_t row = (size_t)(mblk + mloc);
#pragma unroll
            for (int t = 0; t < 8; ++t) {
                const int n = wn * 128 + t * 16 + r;
                out[row * HIDDEN + n] = acc[t][mt][v] * s * norm_w[n];
            }
        }
    }
}

// ---------------------------------------------------------------------------
extern "C" void kernel_launch(void* const* d_in, const int* in_sizes, int n_in,
                              void* d_out, int out_size, void* d_ws, size_t ws_size,
                              hipStream_t stream) {
    const float* x      = (const float*)d_in[0];   // [2,4096,1024] f32
    const float* conv_w = (const float*)d_in[1];   // [20,1024,1024] f32
    const float* norm_w = (const float*)d_in[2];   // [1024] f32
    float*       out    = (float*)d_out;           // [2,4096,1024] f32

    // Workspace: 16 MB bf16 X, then 2 MB bf16 summed W.
    __bf16* xb = (__bf16*)d_ws;
    __bf16* wb = (__bf16*)((char*)d_ws + (size_t)MTOTAL * HIDDEN * sizeof(__bf16));

    x_to_bf16   <<<(MTOTAL * HIDDEN / 4) / 256, 256, 0, stream>>>(x, xb);
    wsum_to_bf16<<<(HIDDEN * HIDDEN / 4) / 256, 256, 0, stream>>>(conv_w, wb);
    gemm_rms    <<<MTOTAL / 32, 256, NBUF * BUF_BYTES, stream>>>(xb, wb, norm_w, out);
}
